// GroupToGroupConv_53953379172475
// MI455X (gfx1250) — matrix operand order repaired and checked by reading, not compiled, not run on hardware
//
#include <hip/hip_runtime.h>

typedef _Float16 f16;
typedef __attribute__((ext_vector_type(8)))  _Float16 v8h;
typedef __attribute__((ext_vector_type(16))) _Float16 v16h;
typedef __attribute__((ext_vector_type(8)))  float    v8f;

#define B_    16
#define CIN   128   // IN_CH * G
#define COUT  128   // OUT_CH * G
#define HH    128
#define WW    128
#define GRP   4
#define ICPG  32    // IN_CH per group
#define OCPG  32    // OUT_CH per group

#define CPAD        136                 // padded f16 channel stride in LDS (bank spread)
#define XPOS        (3 * 130)           // 3 halo rows x 130 halo cols
#define XLDS_HALVES (XPOS * CPAD)       // 53040 halves
#define WLDS_HALVES (64 * CPAD)         // 8704 halves
#define SMEM_BYTES  ((XLDS_HALVES + WLDS_HALVES) * 2)  // 123488 B < 320KB/WGP

extern "C" __global__ __launch_bounds__(256)
void g2g_conv3x3_wmma_f16(const float* __restrict__ x,
                          const float* __restrict__ weight,
                          float* __restrict__ out)
{
    extern __shared__ f16 smem[];
    f16* __restrict__ xlds = smem;                // [pos 0..389][CPAD]
    f16* __restrict__ wlds = smem + XLDS_HALVES;  // [oc 0..63][CPAD]

    const int tid  = threadIdx.x;
    const int lane = tid & 31;
    const int wid  = tid >> 5;          // wave 0..7
    const int hi16 = lane >> 4;         // lane half (0/1)
    const int l16  = lane & 15;

    const int ochalf = blockIdx.x & 1;
    const int h      = (blockIdx.x >> 1) & (HH - 1);
    const int b      = blockIdx.x >> 8;
    const int oc_base = ochalf * 64;

    const int wave_oc = (wid & 1) * 32;   // 0 / 32 within the 64-oc tile
    const int wave_sp = (wid >> 1) * 32;  // 0 / 32 / 64 / 96 within the W row

    // ---- Stage input halo: rows h-1..h+1, cols -1..128, all 128 channels, as f16 ----
    for (int rc = wid; rc < 3 * CIN; rc += 8) {
        const int r   = rc >> 7;                 // 0..2
        const int c   = rc & (CIN - 1);
        const int row = h + r - 1;
        const bool rok = (unsigned)row < (unsigned)HH;
        const float* src = x + (((size_t)b * CIN + c) * HH + (rok ? row : 0)) * WW;
        for (int wp = lane; wp < 130; wp += 32) {     // halo col; w = wp-1
            const int wcol = wp - 1;
            float v = 0.0f;
            if (rok && (unsigned)wcol < (unsigned)WW) v = src[wcol];
            xlds[(r * 130 + wp) * CPAD + c] = (f16)v;
        }
    }

    v8f acc[2][2] = {};   // 32x32 C tile per wave: 2x2 WMMA tiles

    for (int t = 0; t < 9; ++t) {
        const int kh = t / 3, kw = t - kh * 3;

        __syncthreads();  // previous tap finished reading wlds
        // ---- Stage expanded weight tile for this tap: wlds[oc][ic*4+gi] ----
        for (int i = tid; i < 64 * CIN; i += 256) {
            const int oc  = i >> 7;            // 0..63
            const int c   = i & (CIN - 1);     // ic*4 + gi
            const int o   = oc_base + oc;
            const int go  = o >> 5, ocm = o & (OCPG - 1);
            const int ic  = c >> 2, gi  = c & 3;
            const int g   = (go - gi) & 3;
            const float wv = weight[(((ocm * ICPG + ic) * GRP + g) * 9) + t];
            wlds[oc * CPAD + c] = (f16)wv;
        }
        __syncthreads();

        // ---- K loop: 4 chunks of 32 input channels ----
        for (int kc = 0; kc < 4; ++kc) {
            const int kb = kc * 32;
            union { v16h v; v8h h8[2]; } afr[2], bfr[2];

            // A (16x32 f16, M=outch): lanes 0-15 -> K {kb..+7, kb+16..+23},
            // lanes 16-31 -> K {kb+8..+15, kb+24..+31}
            const int ka = kb + (hi16 ? 8 : 0);
#pragma unroll
            for (int i = 0; i < 2; ++i) {
                const f16* s = &wlds[(wave_oc + i * 16 + l16) * CPAD + ka];
                afr[i].h8[0] = *(const v8h*)(s);
                afr[i].h8[1] = *(const v8h*)(s + 16);
            }
            // B (32x16 f16, N=spatial): lanes 0-15 -> K {kb..+7, kb+16..+23},
            // lanes 16-31 -> K {kb+8..+15, kb+24..+31}
            const int kbx = kb + (hi16 ? 8 : 0);
#pragma unroll
            for (int j = 0; j < 2; ++j) {
                const int pos = kh * 130 + (wave_sp + j * 16 + l16 + kw);
                const f16* s = &xlds[pos * CPAD + kbx];
                bfr[j].h8[0] = *(const v8h*)(s);
                bfr[j].h8[1] = *(const v8h*)(s + 16);
            }
#pragma unroll
            for (int i = 0; i < 2; ++i)
#pragma unroll
                for (int j = 0; j < 2; ++j)
                    acc[i][j] = __builtin_amdgcn_wmma_f32_16x16x32_f16(
                        false, afr[i].v, false, bfr[j].v,
                        (short)0, acc[i][j], false, false);
        }
    }

    // ---- Epilogue: C/D layout -> out[b][oc][h][w] (f32) ----
    const int m_off = hi16 * 8;
#pragma unroll
    for (int i = 0; i < 2; ++i) {
#pragma unroll
        for (int j = 0; j < 2; ++j) {
            const int wcol = wave_sp + j * 16 + l16;
#pragma unroll
            for (int r = 0; r < 8; ++r) {
                const int oc = oc_base + wave_oc + i * 16 + r + m_off;
                out[(((size_t)b * COUT + oc) * HH + h) * WW + wcol] = acc[i][j][r];
            }
        }
    }
}

extern "C" void kernel_launch(void* const* d_in, const int* in_sizes, int n_in,
                              void* d_out, int out_size, void* d_ws, size_t ws_size,
                              hipStream_t stream) {
    (void)in_sizes; (void)n_in; (void)d_ws; (void)ws_size; (void)out_size;
    const float* x = (const float*)d_in[0];       // (16, 128, 128, 128) f32
    const float* w = (const float*)d_in[1];       // (32, 32, 4, 3, 3)   f32
    float* out = (float*)d_out;                   // (16, 128, 128, 128) f32

    // Dynamic LDS > 64KB: raise the per-kernel limit (deterministic, capture-safe)
    hipFuncSetAttribute((const void*)g2g_conv3x3_wmma_f16,
                        hipFuncAttributeMaxDynamicSharedMemorySize, SMEM_BYTES);

    dim3 grid(B_ * HH * 2);   // (b, h, oc-half) = 4096 workgroups
    dim3 block(256);          // 8 wave32
    g2g_conv3x3_wmma_f16<<<grid, block, SMEM_BYTES, stream>>>(x, w, out);
}
